// SelfAttention_20409684590662
// MI455X (gfx1250) — matrix operand order repaired and checked by reading, not compiled, not run on hardware
//
#include <hip/hip_runtime.h>
#include <hip/hip_bf16.h>
#include <cstdint>
#include <cstddef>

#define DIM   2048
#define NH    32
#define HD    64
#define BATCH 4
#define SEQ   1025
#define SPAD  1040               // SEQ padded to multiple of 16
#define MPAD  (BATCH * SPAD)     // 4160 rows (multiple of 64)
#define BH    (BATCH * NH)       // 128 (batch, head) pairs
#define NQT   (SPAD / 16)        // 65 query tiles per (b,h)
#define KCH   64                 // K-chunk staged per TDM transfer

typedef __attribute__((ext_vector_type(16))) __bf16       bf16x16;
typedef __attribute__((ext_vector_type(8)))  float        f32x8;
typedef __attribute__((ext_vector_type(4)))  unsigned int u32x4;
typedef __attribute__((ext_vector_type(4)))  unsigned int ui32x4;
typedef __attribute__((ext_vector_type(8)))  int          i32x8;
typedef __attribute__((ext_vector_type(4)))  int          i32x4;

union FragU {
  bf16x16      v;
  u32x4        u[2];
  unsigned int w[8];
};

__device__ __forceinline__ unsigned short f2bf(float f) {
  unsigned int u = __float_as_uint(f);
  u += 0x7FFFu + ((u >> 16) & 1u);      // round-to-nearest-even
  return (unsigned short)(u >> 16);
}
__device__ __forceinline__ float bf2f(unsigned short h) {
  return __uint_as_float((unsigned int)h << 16);
}

__device__ __forceinline__ f32x8 wmma_bf16(bf16x16 a, bf16x16 b, f32x8 c) {
  // D = A(16x32 bf16) * B(32x16 bf16) + C(16x16 f32)
  return __builtin_amdgcn_wmma_f32_16x16x32_bf16(false, a, false, b, (short)0, c,
                                                 false, false);
}

// Byte offset of a __shared__ object inside the workgroup LDS allocation.
__device__ __forceinline__ unsigned lds_byte_of(const void* p) {
  return (unsigned)(unsigned long long)(const __attribute__((address_space(3))) void*)p;
}

// Issue a TDM 2-D tile load: tile_rows x tile_k bf16 elements, row stride
// ld_elems, from global 'gaddr' (tile start) into LDS byte offset 'lds_byte'.
// D# bitfields per CDNA5 ISA ch.8 (group0: count/lds/global/type, group1:
// data_size + dims + strides). Groups 2/3 zero (2-D tensor, no cluster).
__device__ __forceinline__ void tdm_load_2d(unsigned lds_byte, const void* gaddr,
                                            unsigned tile_k, unsigned tile_rows,
                                            unsigned ld_elems) {
  unsigned long long ga = (unsigned long long)gaddr;
  ui32x4 g0;
  g0[0] = 1u;                                              // count=1 (valid user D#)
  g0[1] = lds_byte;                                        // lds_addr
  g0[2] = (unsigned)(ga & 0xFFFFFFFFu);                    // global_addr[31:0]
  g0[3] = (unsigned)((ga >> 32) & 0x01FFFFFFu) | (2u << 30); // global_addr[56:32] | type=2
  i32x8 g1;
  g1[0] = (int)(1u << 16);                                 // data_size=1 (2 bytes)
  g1[1] = (int)(tile_k << 16);                             // tensor_dim0[15:0] @ bits63:48
  g1[2] = (int)((tile_k >> 16) | (tile_rows << 16));       // tensor_dim0 hi | tensor_dim1 lo
  g1[3] = (int)((tile_rows >> 16) | (tile_k << 16));       // tensor_dim1 hi | tile_dim0
  g1[4] = (int)(tile_rows & 0xFFFFu);                      // tile_dim1 (tile_dim2=0)
  g1[5] = (int)ld_elems;                                   // tensor_dim0_stride[31:0]
  g1[6] = 0;                                               // stride hi | dim1_stride lo
  g1[7] = 0;
  i32x4 z4 = {0, 0, 0, 0};
  i32x8 z8 = {0, 0, 0, 0, 0, 0, 0, 0};
  __builtin_amdgcn_tensor_load_to_lds(g0, g1, z4, z4, z8, 0);  // 6-arg form
}

// A-operand fragment (16x32, bf16) from row-major [rows x ld] source.
// lane l: row = base_row + (l&15); K elems {hl*8..hl*8+7} and {16+hl*8..16+hl*8+7}
__device__ __forceinline__ bf16x16 load_a_frag(const unsigned short* base, int ld,
                                               int row, int k0, int lane) {
  int hl = lane >> 4;
  const unsigned short* p =
      base + (size_t)(row + (lane & 15)) * (size_t)ld + (size_t)(k0 + hl * 8);
  FragU f;
  f.u[0] = *(const u32x4*)(p);
  f.u[1] = *(const u32x4*)(p + 16);
  return f.v;
}

// B-operand fragment (K=32 x N=16, bf16) built from a row-major [N x K] source
// (i.e. computing A * W^T). lane l: W row = base_row + (l&15);
// K elems {hl*8..hl*8+7} and {16+hl*8..16+hl*8+7}.
__device__ __forceinline__ bf16x16 load_b_frag(const unsigned short* base, int ld,
                                               int row, int k0, int lane) {
  int hl = lane >> 4;
  const unsigned short* p =
      base + (size_t)(row + (lane & 15)) * (size_t)ld + (size_t)(k0 + hl * 8);
  FragU f;
  f.u[0] = *(const u32x4*)(p);
  f.u[1] = *(const u32x4*)(p + 16);
  return f.v;
}

// ---------------------------------------------------------------- fp32 -> bf16
__global__ void cvt_bf16_kernel(const float* __restrict__ src,
                                unsigned short* __restrict__ dst, long long n) {
  long long i = (long long)blockIdx.x * blockDim.x + threadIdx.x;
  long long stride = (long long)gridDim.x * blockDim.x;
  for (; i < n; i += stride) dst[i] = f2bf(src[i]);
}

// ------------------------------------------------------------------- RMS norm
__global__ void rmsnorm_kernel(const float* __restrict__ x,
                               const float* __restrict__ scale,
                               unsigned short* __restrict__ hb) {
  int row = blockIdx.x;                 // 0 .. MPAD-1
  int b = row / SPAD, s = row % SPAD;
  int tid = threadIdx.x;
  unsigned short* hrow = hb + (size_t)row * DIM;
  if (s >= SEQ) {
    for (int i = tid; i < DIM; i += 256) hrow[i] = 0;
    return;
  }
  const float* xrow = x + ((size_t)(b * SEQ + s)) * DIM;
  float vals[8];
  float ss = 0.f;
#pragma unroll
  for (int i = 0; i < 8; ++i) {
    float v = xrow[tid + i * 256];
    vals[i] = v;
    ss += v * v;
  }
  __shared__ float red[256];
  red[tid] = ss;
  __syncthreads();
#pragma unroll
  for (int off = 128; off > 0; off >>= 1) {
    if (tid < off) red[tid] += red[tid + off];
    __syncthreads();
  }
  float rs = rsqrtf(red[0] * (1.0f / (float)DIM) + 1e-6f);
#pragma unroll
  for (int i = 0; i < 8; ++i)
    hrow[tid + i * 256] = f2bf(vals[i] * scale[tid + i * 256] * rs);
}

// --------------------------------------------------- WMMA GEMM: C = A * W^T
// A: [MPAD x K] bf16 row-major (64-row block tile staged via TDM into LDS,
// double buffered, consumed by all 8 waves). W: [N x K] bf16 row-major,
// direct-global (L2 resident). B fragments for a k-step are hoisted into
// distinct registers so the 8 global_load_b128 can clause-batch and the
// s_wait_loadcnt waits stagger across the WMMA burst.
// mode 0: store bf16 into outb [MPAD x N].
// mode 1: fp32 out with residual add, rows remapped (b*SPAD+s) -> (b*SEQ+s).
__global__ void __launch_bounds__(256) gemm_bf16_wmma(
    const unsigned short* __restrict__ A, const unsigned short* __restrict__ W,
    int K, int N, unsigned short* __restrict__ outb, float* __restrict__ outf,
    const float* __restrict__ skip, int mode) {
  __shared__ __align__(16) unsigned short smA[2][64 * KCH];  // 16 KB double buffer

  int lane = threadIdx.x & 31;
  int wid = threadIdx.x >> 5;           // 8 waves
  int wm = wid >> 2;                    // 0..1
  int wn = wid & 3;                     // 0..3
  int rowBase = blockIdx.y * 64;        // block's 64-row A tile
  int colBase = blockIdx.x * 256 + wn * 64;

  const f32x8 zero = {0.f, 0.f, 0.f, 0.f, 0.f, 0.f, 0.f, 0.f};
  f32x8 acc[2][4];
#pragma unroll
  for (int mi = 0; mi < 2; ++mi)
#pragma unroll
    for (int ni = 0; ni < 4; ++ni) acc[mi][ni] = zero;

  const int nkc = K / KCH;
  unsigned ldsA[2] = {lds_byte_of(&smA[0][0]), lds_byte_of(&smA[1][0])};
  const bool issuer = (threadIdx.x < 32);     // wave 0 drives the TDM pipeline

  if (issuer) {
    tdm_load_2d(ldsA[0], A + (size_t)rowBase * K, KCH, 64, (unsigned)K);
    tdm_load_2d(ldsA[1], A + (size_t)rowBase * K + KCH, KCH, 64, (unsigned)K);
  }

  for (int kc = 0; kc < nkc; ++kc) {
    if (issuer) {
      if (kc + 1 < nkc) __builtin_amdgcn_s_wait_tensorcnt((short)1);
      else              __builtin_amdgcn_s_wait_tensorcnt((short)0);
    }
    __syncthreads();                    // chunk kc resident in smA[kc&1]

    const unsigned short* At = &smA[kc & 1][0];
#pragma unroll
    for (int kk = 0; kk < KCH; kk += 32) {
      int kg = kc * KCH + kk;
      bf16x16 bfr[4];
#pragma unroll
      for (int ni = 0; ni < 4; ++ni)
        bfr[ni] = load_b_frag(W, K, colBase + ni * 16, kg, lane);
      bf16x16 a0 = load_a_frag(At, KCH, wm * 32, kk, lane);
      bf16x16 a1 = load_a_frag(At, KCH, wm * 32 + 16, kk, lane);
#pragma unroll
      for (int ni = 0; ni < 4; ++ni) {
        acc[0][ni] = wmma_bf16(a0, bfr[ni], acc[0][ni]);
        acc[1][ni] = wmma_bf16(a1, bfr[ni], acc[1][ni]);
      }
    }
    __syncthreads();                    // everyone done with smA[kc&1]
    if (issuer && (kc + 2 < nkc)) {
      tdm_load_2d(ldsA[kc & 1], A + (size_t)rowBase * K + (size_t)(kc + 2) * KCH,
                  KCH, 64, (unsigned)K);
    }
  }

  int hl = lane >> 4;
  int cl = lane & 15;
#pragma unroll
  for (int mi = 0; mi < 2; ++mi) {
#pragma unroll
    for (int v = 0; v < 8; ++v) {
      int r = rowBase + wm * 32 + mi * 16 + v + 8 * hl;
      if (mode == 0) {
#pragma unroll
        for (int ni = 0; ni < 4; ++ni) {
          int c = colBase + ni * 16 + cl;
          outb[(size_t)r * N + c] = f2bf(acc[mi][ni][v]);
        }
      } else {
        int b = r / SPAD, s = r % SPAD;
        if (s < SEQ) {
          size_t rowo = ((size_t)(b * SEQ + s)) * (size_t)N;
#pragma unroll
          for (int ni = 0; ni < 4; ++ni) {
            int c = colBase + ni * 16 + cl;
            outf[rowo + c] = acc[mi][ni][v] + skip[rowo + c];
          }
        }
      }
    }
  }
}

// ------------------------------------------------------- RoPE + q/k/v scatter
__global__ void rope_kernel(const unsigned short* __restrict__ qkvb,
                            unsigned short* __restrict__ q_out,
                            unsigned short* __restrict__ k_out,
                            unsigned short* __restrict__ vt_out) {
  int t = threadIdx.x & 31;             // rotary pair lane: d and d+32
  int w = threadIdx.x >> 5;
  int sg = blockIdx.x % (SPAD / 8);
  int bh = blockIdx.x / (SPAD / 8);     // b*NH + h
  int h = bh % NH, b = bh / NH;
  int s = sg * 8 + w;

  size_t qrow = ((size_t)bh * SPAD + s) * HD;
  size_t vbase = (size_t)bh * HD * SPAD;
  if (s >= SEQ) {
    q_out[qrow + t] = 0; q_out[qrow + t + 32] = 0;
    k_out[qrow + t] = 0; k_out[qrow + t + 32] = 0;
    vt_out[vbase + (size_t)t * SPAD + s] = 0;
    vt_out[vbase + (size_t)(t + 32) * SPAD + s] = 0;
    return;
  }
  size_t xrow = ((size_t)b * SPAD + s) * (size_t)(3 * DIM);
  int n0 = h * HD + t;
  float q1 = bf2f(qkvb[xrow + n0]);
  float q2 = bf2f(qkvb[xrow + n0 + 32]);
  float k1 = bf2f(qkvb[xrow + DIM + n0]);
  float k2 = bf2f(qkvb[xrow + DIM + n0 + 32]);
  float v1 = bf2f(qkvb[xrow + 2 * DIM + n0]);
  float v2 = bf2f(qkvb[xrow + 2 * DIM + n0 + 32]);

  float fidx = (float)(t & 15);
  float freq = 3.14159265358979f * __expf(2.302585092994046f * (fidx * (1.0f / 15.0f)));
  float pv = 0.f;
  if (s > 0) {
    int p = s - 1;
    int pi = p >> 5, pj = p & 31;
    float hp = (float)(2 * pi + 1) * (1.0f / 32.0f) - 1.0f;
    float wp = (float)(2 * pj + 1) * (1.0f / 32.0f) - 1.0f;
    pv = (t < 16) ? hp : wp;
  }
  float th = pv * freq;
  float sn, cs;
  __sincosf(th, &sn, &cs);

  float qa = q1 * cs - q2 * sn, qbv = q2 * cs + q1 * sn;
  float ka = k1 * cs - k2 * sn, kbv = k2 * cs + k1 * sn;
  const float qscale = 0.125f;          // 1/sqrt(HEAD_DIM)
  q_out[qrow + t]      = f2bf(qa * qscale);
  q_out[qrow + t + 32] = f2bf(qbv * qscale);
  k_out[qrow + t]      = f2bf(ka);
  k_out[qrow + t + 32] = f2bf(kbv);
  vt_out[vbase + (size_t)t * SPAD + s]        = f2bf(v1);
  vt_out[vbase + (size_t)(t + 32) * SPAD + s] = f2bf(v2);
}

// ----------------------------------------------------- flash attention (wave)
// One wave per (b, h, 16-query tile). Scores computed transposed
// (St = K*Q^T) so softmax lanes == queries; Pt converts to the P*V B-operand
// in-lane (shfl_xor results unused); O accumulated transposed (Ot = Vt * Pt).
__global__ void __launch_bounds__(32) attn_kernel(
    const unsigned short* __restrict__ qb, const unsigned short* __restrict__ kb,
    const unsigned short* __restrict__ vt, unsigned short* __restrict__ o) {
  int lane = threadIdx.x;
  int qt = blockIdx.x % NQT;
  int bh = blockIdx.x / NQT;            // b*NH + h
  int h = bh % NH, b = bh / NH;

  const unsigned short* Q  = qb + (size_t)bh * SPAD * HD;
  const unsigned short* Kp = kb + (size_t)bh * SPAD * HD;
  const unsigned short* Vt = vt + (size_t)bh * HD * SPAD;

  int hl = lane >> 4;
  int cl = lane & 15;
  int sq = qt * 16 + cl;                // this lane's query position

  bf16x16 qf[2];
  qf[0] = load_b_frag(Q, HD, qt * 16, 0, lane);
  qf[1] = load_b_frag(Q, HD, qt * 16, 32, lane);

  const f32x8 zero = {0.f, 0.f, 0.f, 0.f, 0.f, 0.f, 0.f, 0.f};
  f32x8 ot[4];
#pragma unroll
  for (int f = 0; f < 4; ++f) ot[f] = zero;
  float m_run = -3.0e38f, l_run = 0.f;

  int nblk = (qt * 16 + 16 + 31) >> 5;  // 32-key blocks covering causal span
  for (int kbi = 0; kbi < nblk; ++kbi) {
    int jb = kbi * 32;
    f32x8 st0 = zero, st1 = zero;
#pragma unroll
    for (int dc = 0; dc < 2; ++dc) {
      bf16x16 a0 = load_a_frag(Kp, HD, jb, dc * 32, lane);
      bf16x16 a1 = load_a_frag(Kp, HD, jb + 16, dc * 32, lane);
      st0 = wmma_bf16(a0, qf[dc], st0);
      st1 = wmma_bf16(a1, qf[dc], st1);
    }
    float p0[8], p1[8];
    float mx = -3.0e38f;
#pragma unroll
    for (int v = 0; v < 8; ++v) {
      int j0 = jb + v + 8 * hl;
      int j1 = j0 + 16;
      p0[v] = (j0 <= sq) ? st0[v] : -3.0e38f;
      p1[v] = (j1 <= sq) ? st1[v] : -3.0e38f;
      mx = fmaxf(mx, fmaxf(p0[v], p1[v]));
    }
    mx = fmaxf(mx, __shfl_xor(mx, 16));
    float m_new = fmaxf(m_run, mx);
    float corr = __expf(m_run - m_new);
    float lsum = 0.f;
#pragma unroll
    for (int v = 0; v < 8; ++v) {
      p0[v] = (p0[v] > -1.0e30f) ? __expf(p0[v] - m_new) : 0.f;
      p1[v] = (p1[v] > -1.0e30f) ? __expf(p1[v] - m_new) : 0.f;
      lsum += p0[v] + p1[v];
    }
    lsum += __shfl_xor(lsum, 16);
    l_run = l_run * corr + lsum;
    m_run = m_new;

    unsigned int pw0[4], pw1[4];
#pragma unroll
    for (int i = 0; i < 4; ++i) {
      pw0[i] = (unsigned int)f2bf(p0[2 * i]) |
               ((unsigned int)f2bf(p0[2 * i + 1]) << 16);
      pw1[i] = (unsigned int)f2bf(p1[2 * i]) |
               ((unsigned int)f2bf(p1[2 * i + 1]) << 16);
    }
    FragU pb;
#pragma unroll
    for (int i = 0; i < 4; ++i) {
      unsigned int x0 = __shfl_xor(pw0[i], 16);
      unsigned int x1 = __shfl_xor(pw1[i], 16);
      pb.w[i]     = pw0[i];   // key offsets  0..7  / 8..15
      pb.w[4 + i] = pw1[i];   // key offsets  16..23 / 24..31
    }
#pragma unroll
    for (int f = 0; f < 4; ++f) ot[f] = ot[f] * corr;
#pragma unroll
    for (int f = 0; f < 4; ++f) {
      bf16x16 av = load_a_frag(Vt, SPAD, f * 16, jb, lane);
      ot[f] = wmma_bf16(av, pb.v, ot[f]);
    }
  }

  float inv = 1.0f / l_run;
  unsigned short* orow =
      o + ((size_t)b * SPAD + (size_t)(qt * 16 + cl)) * DIM + (size_t)h * HD;
#pragma unroll
  for (int f = 0; f < 4; ++f)
#pragma unroll
    for (int v = 0; v < 8; ++v)
      orow[f * 16 + v + 8 * hl] = f2bf(ot[f][v] * inv);
}

// ------------------------------------------------------------------ launcher
extern "C" void kernel_launch(void* const* d_in, const int* in_sizes, int n_in,
                              void* d_out, int out_size, void* d_ws, size_t ws_size,
                              hipStream_t stream) {
  (void)in_sizes; (void)n_in; (void)out_size; (void)ws_size;
  const float* x      = (const float*)d_in[0];
  const float* scale  = (const float*)d_in[1];
  const float* w_qkv  = (const float*)d_in[2];
  const float* w_out  = (const float*)d_in[3];
  float* out = (float*)d_out;

  char* ws = (char*)d_ws;
  size_t off = 0;
  auto take = [&](size_t bytes) -> char* {
    char* p = ws + off;
    off += (bytes + 255) & ~(size_t)255;
    return p;
  };
  unsigned short* wqkv_b = (unsigned short*)take((size_t)3 * DIM * DIM * 2);
  unsigned short* wout_b = (unsigned short*)take((size_t)DIM * DIM * 2);
  unsigned short* hb     = (unsigned short*)take((size_t)MPAD * DIM * 2);   // reused as o
  unsigned short* qkvb   = (unsigned short*)take((size_t)MPAD * 3 * DIM * 2);
  unsigned short* qbuf   = (unsigned short*)take((size_t)BH * SPAD * HD * 2);
  unsigned short* kbuf   = (unsigned short*)take((size_t)BH * SPAD * HD * 2);
  unsigned short* vtbuf  = (unsigned short*)take((size_t)BH * HD * SPAD * 2);

  cvt_bf16_kernel<<<4096, 256, 0, stream>>>(w_qkv, wqkv_b, (long long)3 * DIM * DIM);
  cvt_bf16_kernel<<<2048, 256, 0, stream>>>(w_out, wout_b, (long long)DIM * DIM);

  rmsnorm_kernel<<<MPAD, 256, 0, stream>>>(x, scale, hb);

  gemm_bf16_wmma<<<dim3((3 * DIM) / 256, MPAD / 64), 256, 0, stream>>>(
      hb, wqkv_b, DIM, 3 * DIM, qkvb, nullptr, nullptr, 0);

  rope_kernel<<<BH * (SPAD / 8), 256, 0, stream>>>(qkvb, qbuf, kbuf, vtbuf);

  attn_kernel<<<BH * NQT, 32, 0, stream>>>(qbuf, kbuf, vtbuf, hb /* o */);

  gemm_bf16_wmma<<<dim3(DIM / 256, MPAD / 64), 256, 0, stream>>>(
      hb, wout_b, DIM, DIM, nullptr, out, x, 1);
}
